// CNNTransMIL_66872640798854
// MI455X (gfx1250) — matrix operand order repaired and checked by reading, not compiled, not run on hardware
//
#include <hip/hip_runtime.h>
#include <hip/hip_bf16.h>

typedef __attribute__((ext_vector_type(16))) __bf16        v16bf;
typedef __attribute__((ext_vector_type(8)))  float         v8f;
typedef __attribute__((ext_vector_type(8)))  unsigned int  v8ui;

// pack two f32 into one dword holding two bf16 (round-to-nearest-even)
__device__ __forceinline__ unsigned int pack2bf(float a, float b) {
  unsigned int ua = __builtin_bit_cast(unsigned int, a);
  unsigned int ub = __builtin_bit_cast(unsigned int, b);
  ua = ua + 0x7FFFu + ((ua >> 16) & 1u);
  ub = ub + 0x7FFFu + ((ub >> 16) & 1u);
  return (ua >> 16) | (ub & 0xFFFF0000u);
}

// ---------------------------------------------------------------------------
// Generic WMMA GEMM: C = epilogue( alpha * A @ op(B) )  (bf16 inputs, f32 acc)
//   block: 128 threads (4 wave32); tile 128(M) x 64(N); K step 32
//   each wave: 32 rows (2 A fragments) x 64 cols (4 B fragments) = 8 WMMA/step
//   transB==0: B given as W[n][k] (k contiguous)  -> C = A @ W^T
//   transB==1: B given as B[k][n] (n contiguous)  -> C = A @ B
//   a_mode==1: patch-embed swizzle A[m][k] = x[m*16384 + (k&4095)*4 + (k>>12)]
//   epilogue: v = alpha*acc (+ diagv on i==j) (+ bias[col]) (relu) (+= C)
//   row_mode==1: out row = m + m/2047 + 1  (fc1 -> token slot, skipping cls)
// ---------------------------------------------------------------------------
__global__ __launch_bounds__(128) void k_gemm(
    const float* __restrict__ A, const float* __restrict__ Bm,
    float* __restrict__ C, const float* __restrict__ bias,
    int M, int N, int K, int lda, int ldb, int ldc,
    long long sAb, long long sAh, long long sBb, long long sBh,
    long long sCb, long long sCh, int nH,
    int transB, int a_mode, int relu, int accum, int row_mode,
    float alpha, float diagv)
{
  // B tile staged as packed bf16 pairs: bt[n][kpair], kpair j holds k=2j,2j+1
  __shared__ unsigned int bt[64][17];

  int z  = blockIdx.z;
  int bb = z / nH, hh = z % nH;
  const float* Ab = A  + (size_t)bb * sAb + (size_t)hh * sAh;
  const float* Bb = Bm + (size_t)bb * sBb + (size_t)hh * sBh;
  float*       Cb = C  + (size_t)bb * sCb + (size_t)hh * sCh;

  int tid  = threadIdx.x;
  int wid  = tid >> 5, lane = tid & 31;
  int half = lane >> 4, l16 = lane & 15;
  int rowBase = blockIdx.y * 128 + wid * 32;
  int colBase = blockIdx.x * 64;

  v8f acc[2][4] = {};

  int aRow[2];
  aRow[0] = rowBase + l16;       if (aRow[0] >= M) aRow[0] = M - 1;
  aRow[1] = rowBase + 16 + l16;  if (aRow[1] >= M) aRow[1] = M - 1;

  for (int k0 = 0; k0 < K; k0 += 32) {
    // ---- stage B tile (32 k x 64 n) into LDS as packed bf16 ----
    if (!transB) {
      int n  = tid >> 1;                   // 0..63
      int kh = (tid & 1) * 16;             // 0 or 16
      const float* src = Bb + (size_t)(colBase + n) * (size_t)ldb + (size_t)(k0 + kh);
      #pragma unroll
      for (int j = 0; j < 8; ++j)
        bt[n][(kh >> 1) + j] = pack2bf(src[2 * j], src[2 * j + 1]);
    } else {
      int kp = tid >> 3;                   // k pair 0..15 -> k = 2kp, 2kp+1
      int ns = (tid & 7) * 8;              // 8 columns per thread
      const float* r0 = Bb + (size_t)(k0 + 2 * kp) * (size_t)ldb + (size_t)(colBase + ns);
      const float* r1 = r0 + (size_t)ldb;
      #pragma unroll
      for (int e = 0; e < 8; ++e)
        bt[ns + e][kp] = pack2bf(r0[e], r1[e]);
    }
    __syncthreads();

    // ---- A fragments (2 x 16x32), per documented 16-bit A layout ----
    v16bf af[2];
    #pragma unroll
    for (int rf = 0; rf < 2; ++rf) {
      v8ui au;
      if (a_mode == 1) {
        #pragma unroll
        for (int j = 0; j < 8; ++j) {
          int kk = k0 + ((j >> 2) << 4) + (half << 3) + ((j & 3) << 1);
          float f0 = A[(size_t)aRow[rf] * 16384 + ((size_t)(kk & 4095) << 2) + (size_t)(kk >> 12)];
          int k1 = kk + 1;
          float f1 = A[(size_t)aRow[rf] * 16384 + ((size_t)(k1 & 4095) << 2) + (size_t)(k1 >> 12)];
          au[j] = pack2bf(f0, f1);
        }
      } else {
        const float* ar = Ab + (size_t)aRow[rf] * (size_t)lda + (size_t)k0;
        #pragma unroll
        for (int j = 0; j < 8; ++j) {
          int kk = ((j >> 2) << 4) + (half << 3) + ((j & 3) << 1);
          au[j] = pack2bf(ar[kk], ar[kk + 1]);
        }
      }
      af[rf] = __builtin_bit_cast(v16bf, au);
    }

    // ---- 4 column fragments x 2 row fragments -> 8 WMMA ----
    #pragma unroll
    for (int f = 0; f < 4; ++f) {
      v8ui bu;
      int nc = f * 16 + l16;
      #pragma unroll
      for (int j = 0; j < 8; ++j) bu[j] = bt[nc][((j >> 2) << 3) + half * 4 + (j & 3)];
      v16bf bf = __builtin_bit_cast(v16bf, bu);
      acc[0][f] = __builtin_amdgcn_wmma_f32_16x16x32_bf16(
          false, af[0], false, bf, (short)0, acc[0][f], false, false);
      acc[1][f] = __builtin_amdgcn_wmma_f32_16x16x32_bf16(
          false, af[1], false, bf, (short)0, acc[1][f], false, false);
    }
    __syncthreads();
  }

  // ---- epilogue ----
  #pragma unroll
  for (int rf = 0; rf < 2; ++rf) {
    #pragma unroll
    for (int f = 0; f < 4; ++f) {
      int col = colBase + f * 16 + l16;
      #pragma unroll
      for (int r = 0; r < 8; ++r) {
        int gm = rowBase + rf * 16 + half * 8 + r;
        if (gm >= M) continue;
        float v = acc[rf][f][r] * alpha;
        if (diagv != 0.f && gm == col) v += diagv;
        if (bias) v += bias[col];
        if (relu) v = fmaxf(v, 0.f);
        long long orow = gm;
        if (row_mode == 1) orow = gm + gm / 2047 + 1;
        float* dst = Cb + (size_t)orow * (size_t)ldc + (size_t)col;
        if (accum) v += *dst;
        *dst = v;
      }
    }
  }
}

// ---------------------------------------------------------------------------
// Elementwise / reduction kernels
// ---------------------------------------------------------------------------
__global__ __launch_bounds__(256) void k_layernorm(
    const float* __restrict__ in, float* __restrict__ out,
    const float* __restrict__ g, const float* __restrict__ b, int dim)
{
  __shared__ float red[256];
  int row = blockIdx.x, tid = threadIdx.x;
  const float* R = in + (size_t)row * dim;
  float s = 0.f;
  for (int c = tid; c < dim; c += 256) s += R[c];
  red[tid] = s; __syncthreads();
  for (int st = 128; st > 0; st >>= 1) { if (tid < st) red[tid] += red[tid + st]; __syncthreads(); }
  float mu = red[0] / dim; __syncthreads();
  float v = 0.f;
  for (int c = tid; c < dim; c += 256) { float d = R[c] - mu; v += d * d; }
  red[tid] = v; __syncthreads();
  for (int st = 128; st > 0; st >>= 1) { if (tid < st) red[tid] += red[tid + st]; __syncthreads(); }
  float rstd = rsqrtf(red[0] / dim + 1e-5f); __syncthreads();
  float* O = out + (size_t)row * dim;
  for (int c = tid; c < dim; c += 256) O[c] = (R[c] - mu) * rstd * g[c] + b[c];
}

__global__ __launch_bounds__(256) void k_softmax(float* __restrict__ X, int ncols)
{
  __shared__ float red[256];
  int tid = threadIdx.x;
  float* R = X + (size_t)blockIdx.x * ncols;
  float m = -3.4e38f;
  for (int c = tid; c < ncols; c += 256) m = fmaxf(m, R[c]);
  red[tid] = m; __syncthreads();
  for (int st = 128; st > 0; st >>= 1) { if (tid < st) red[tid] = fmaxf(red[tid], red[tid + st]); __syncthreads(); }
  m = red[0]; __syncthreads();
  float s = 0.f;
  for (int c = tid; c < ncols; c += 256) { float e = expf(R[c] - m); R[c] = e; s += e; }
  red[tid] = s; __syncthreads();
  for (int st = 128; st > 0; st >>= 1) { if (tid < st) red[tid] += red[tid + st]; __syncthreads(); }
  float inv = 1.f / red[0];
  __syncthreads();
  for (int c = tid; c < ncols; c += 256) R[c] *= inv;
}

__global__ __launch_bounds__(256) void k_concat(
    const float* __restrict__ xe, const float* __restrict__ drug,
    const float* __restrict__ hk, float* __restrict__ Hc)
{
  size_t idx = (size_t)blockIdx.x * 256 + threadIdx.x;  // 4094*2560 exact
  int k = (int)(idx % 2560);
  size_t m = idx / 2560;
  int b = (int)(m / 2047);
  float v;
  if (k < 1536)      v = xe[m * 1536 + k];
  else if (k < 2048) v = drug[b * 512 + (k - 1536)];
  else               v = hk[b * 512 + (k - 2048)];
  Hc[idx] = v;
}

__global__ void k_set_cls(const float* __restrict__ cls, float* __restrict__ H)
{
  int idx = blockIdx.x * 256 + threadIdx.x;   // 2048
  int b = idx >> 10, d = idx & 1023;
  H[(size_t)b * 2048 * 1024 + d] = cls[d];
}

__global__ __launch_bounds__(256) void k_scale_q(float* __restrict__ qkv)
{
  size_t idx = (size_t)blockIdx.x * 256 + threadIdx.x;  // 2*2048*1024
  size_t row = idx >> 10;
  int c = (int)(idx & 1023);
  qkv[row * 3072 + c] *= 0.08838834764831845f;          // 128^-0.5
}

__global__ __launch_bounds__(256) void k_landmark(
    const float* __restrict__ qkv, float* __restrict__ out, int coloff)
{
  int idx = blockIdx.x * 256 + threadIdx.x;  // 2*8*512*128 = 1M
  int d = idx & 127, i = (idx >> 7) & 511, h = (idx >> 16) & 7, b = idx >> 19;
  const float* base = qkv + ((size_t)b * 2048 + (size_t)i * 4) * 3072 + coloff + h * 128 + d;
  float s = base[0] + base[3072] + base[2 * 3072] + base[3 * 3072];
  out[idx] = s * 0.25f;
}

__global__ __launch_bounds__(256) void k_resconv(
    const float* __restrict__ qkv, const float* __restrict__ rw, float* __restrict__ att)
{
  int idx = blockIdx.x * 256 + threadIdx.x;  // 2*2048*1024
  int c = idx & 1023, bt = idx >> 10, t = bt & 2047, b = bt >> 11;
  int h = c >> 7;
  const float* v = qkv + (size_t)b * 2048 * 3072 + 2048 + c;
  float s = 0.f;
  #pragma unroll
  for (int k = 0; k < 33; ++k) {
    int tt = t + k - 16;
    if (tt >= 0 && tt < 2048) s += v[(size_t)tt * 3072] * rw[h * 33 + k];
  }
  att[idx] += s;
}

__global__ __launch_bounds__(256) void k_ppeg(
    const float* __restrict__ in, float* __restrict__ out,
    const float* __restrict__ w, const float* __restrict__ bias, int ksz)
{
  int idx = blockIdx.x * 256 + threadIdx.x;  // 2*2048*1024
  int c = idx & 1023, bt = idx >> 10, t = bt & 2047, b = bt >> 11;
  const float* I = in + (size_t)b * 2048 * 1024;
  float x = I[(size_t)t * 1024 + c];
  if (t == 0) { out[idx] = x; return; }
  int i = t - 1, pad = ksz >> 1;
  float s = 0.f;
  for (int k = 0; k < ksz; ++k) {
    int ii = i + k - pad;
    if (ii >= 0 && ii < 2047) s += I[(size_t)(1 + ii) * 1024 + c] * w[c * ksz + k];
  }
  out[idx] = x + s + bias[c];
}

__global__ __launch_bounds__(256) void k_pinv_rowcol(
    const float* __restrict__ a2, float* __restrict__ buf)
{
  __shared__ float red[256];
  int bh = blockIdx.x, tid = threadIdx.x;
  const float* X = a2 + (size_t)bh * 262144;
  float mr = 0.f;
  for (int r = tid; r < 512; r += 256) {
    float s = 0.f;
    for (int j = 0; j < 512; ++j) s += fabsf(X[(size_t)r * 512 + j]);
    mr = fmaxf(mr, s);
  }
  red[tid] = mr; __syncthreads();
  for (int st = 128; st > 0; st >>= 1) { if (tid < st) red[tid] = fmaxf(red[tid], red[tid + st]); __syncthreads(); }
  if (tid == 0) buf[bh] = red[0];
  __syncthreads();
  float mc = 0.f;
  for (int c = tid; c < 512; c += 256) {
    float s = 0.f;
    for (int j = 0; j < 512; ++j) s += fabsf(X[(size_t)j * 512 + c]);
    mc = fmaxf(mc, s);
  }
  red[tid] = mc; __syncthreads();
  for (int st = 128; st > 0; st >>= 1) { if (tid < st) red[tid] = fmaxf(red[tid], red[tid + st]); __syncthreads(); }
  if (tid == 0) buf[16 + bh] = red[0];
}

__global__ __launch_bounds__(256) void k_pinv_make_z(
    const float* __restrict__ a2, const float* __restrict__ buf, float* __restrict__ z)
{
  int bh = blockIdx.x;
  float s1 = 0.f, s2 = 0.f;
  for (int i = 0; i < 16; ++i) { s1 = fmaxf(s1, buf[i]); s2 = fmaxf(s2, buf[16 + i]); }
  float inv = 1.f / (s1 * s2);                 // global scale across (b,h)
  const float* X = a2 + (size_t)bh * 262144;
  float* Z = z + (size_t)bh * 262144;
  for (int i = threadIdx.x + blockIdx.y * 256; i < 262144; i += 256 * 64) {
    int r = i >> 9, c = i & 511;
    Z[i] = X[(size_t)c * 512 + r] * inv;       // transpose / scale
  }
}

__global__ __launch_bounds__(256) void k_diag_lin(
    const float* __restrict__ P, float* __restrict__ Q, float alpha, float dg)
{
  size_t i = (size_t)blockIdx.x * 256 + threadIdx.x;  // 16*512*512
  int r = (int)((i >> 9) & 511), c = (int)(i & 511);
  Q[i] = alpha * P[i] + ((r == c) ? dg : 0.f);
}

__global__ __launch_bounds__(256) void k_head(
    const float* __restrict__ xl, const float* __restrict__ w,
    const float* __restrict__ bias, float* __restrict__ out)
{
  __shared__ float red[256];
  int b = blockIdx.x >> 1, c = blockIdx.x & 1, tid = threadIdx.x;
  const float* R = xl + (size_t)b * 2048 * 1024;   // token 0 of batch b
  float s = 0.f;
  for (int d = tid; d < 1024; d += 256) s += R[d] * w[c * 1024 + d];
  red[tid] = s; __syncthreads();
  for (int st = 128; st > 0; st >>= 1) { if (tid < st) red[tid] += red[tid + st]; __syncthreads(); }
  if (tid == 0) out[blockIdx.x] = red[0] + bias[c];
}

// ---------------------------------------------------------------------------
static inline void gemm_launch(hipStream_t st,
    const float* A, const float* B, float* C, const float* bias,
    int M, int N, int K, int lda, int ldb, int ldc,
    long long sAb, long long sAh, long long sBb, long long sBh,
    long long sCb, long long sCh, int nH, int nbatch,
    int transB, int a_mode, int relu, int accum, int row_mode,
    float alpha, float diagv)
{
  dim3 g((unsigned)(N / 64), (unsigned)((M + 127) / 128), (unsigned)nbatch);
  k_gemm<<<g, 128, 0, st>>>(A, B, C, bias, M, N, K, lda, ldb, ldc,
                            sAb, sAh, sBb, sBh, sCb, sCh, nH,
                            transB, a_mode, relu, accum, row_mode, alpha, diagv);
}

extern "C" void kernel_launch(void* const* d_in, const int* in_sizes, int n_in,
                              void* d_out, int out_size, void* d_ws, size_t ws_size,
                              hipStream_t stream)
{
  (void)in_sizes; (void)n_in; (void)out_size; (void)ws_size;
  const float* x        = (const float*)d_in[0];
  const float* drug     = (const float*)d_in[1];
  const float* Hkmer    = (const float*)d_in[2];
  const float* patch_w  = (const float*)d_in[3];
  const float* patch_b  = (const float*)d_in[4];
  const float* kmer_g   = (const float*)d_in[5];
  const float* kmer_b   = (const float*)d_in[6];
  const float* fc1_w    = (const float*)d_in[7];
  const float* fc1_b    = (const float*)d_in[8];
  const float* cls_tok  = (const float*)d_in[9];
  const float* ln1_g    = (const float*)d_in[10];
  const float* ln1_b    = (const float*)d_in[11];
  const float* qkv1_w   = (const float*)d_in[12];
  const float* out1_w   = (const float*)d_in[13];
  const float* out1_b   = (const float*)d_in[14];
  const float* res1_w   = (const float*)d_in[15];
  const float* ppeg_w7  = (const float*)d_in[16];
  const float* ppeg_b7  = (const float*)d_in[17];
  const float* ppeg_w5  = (const float*)d_in[18];
  const float* ppeg_b5  = (const float*)d_in[19];
  const float* ppeg_w3  = (const float*)d_in[20];
  const float* ppeg_b3  = (const float*)d_in[21];
  const float* ln2_g    = (const float*)d_in[22];
  const float* ln2_b    = (const float*)d_in[23];
  const float* qkv2_w   = (const float*)d_in[24];
  const float* out2_w   = (const float*)d_in[25];
  const float* out2_b   = (const float*)d_in[26];
  const float* res2_w   = (const float*)d_in[27];
  const float* normf_g  = (const float*)d_in[28];
  const float* normf_b  = (const float*)d_in[29];
  const float* fc2_w    = (const float*)d_in[30];
  const float* fc2_b    = (const float*)d_in[31];

  float* Wp  = (float*)d_ws;
  float* H   = Wp;                      // 2*2048*1024
  float* HKN = H + 4194304;             // 1024
  float* BUF = HKN + 1024;
  // phase A (overlaps phase B region)
  float* XE  = BUF;                     // 4094*1536
  float* HC  = XE + 6288384;            // 4094*2560
  // phase B
  float* XL  = BUF;                     // 4194304
  float* QKV = XL + 4194304;            // 12582912
  float* QL  = QKV + 12582912;          // 1048576
  float* KL  = QL + 1048576;            // 1048576
  float* A1  = KL + 1048576;            // 16777216
  float* A2  = A1 + 16777216;           // 4194304
  float* A3  = A2 + 4194304;            // 16777216
  float* ZB  = A3 + 16777216;           // 4194304
  float* Z2  = ZB + 4194304;            // 4194304
  float* PB  = Z2 + 4194304;            // 4194304
  float* W1  = PB + 4194304;            // 4194304
  float* W2  = W1 + 4194304;            // 4194304
  float* A3V = W2 + 4194304;            // 1048576
  float* T2  = A3V + 1048576;           // 1048576
  float* ATT = T2 + 1048576;            // 4194304
  float* RED = ATT + 4194304;           // 32

  const long long SQB = 2048LL * 3072;
  const long long SLM = 512LL * 128, SLMB = 8 * SLM;
  const long long SA1 = 2048LL * 512, SA1B = 8 * SA1;
  const long long SA2 = 512LL * 512,  SA2B = 8 * SA2;
  const long long SA3 = 512LL * 2048, SA3B = 8 * SA3;

  // ---- embedding ----
  k_layernorm<<<2, 256, 0, stream>>>(Hkmer, HKN, kmer_g, kmer_b, 512);
  gemm_launch(stream, x, patch_w, XE, patch_b, 4094, 1536, 16384, 16384, 16384, 1536,
              0, 0, 0, 0, 0, 0, 1, 1, 0, /*a_mode*/1, 0, 0, 0, 1.f, 0.f);
  k_concat<<<40940, 256, 0, stream>>>(XE, drug, HKN, HC);
  gemm_launch(stream, HC, fc1_w, H, fc1_b, 4094, 1024, 2560, 2560, 2560, 1024,
              0, 0, 0, 0, 0, 0, 1, 1, 0, 0, /*relu*/1, 0, /*row_mode*/1, 1.f, 0.f);
  k_set_cls<<<8, 256, 0, stream>>>(cls_tok, H);

  // ---- nystrom attention layer ----
  auto attn_layer = [&](const float* lng, const float* lnb, const float* qkvw,
                        const float* outw, const float* outb, const float* resw) {
    k_layernorm<<<4096, 256, 0, stream>>>(H, XL, lng, lnb, 1024);
    gemm_launch(stream, XL, qkvw, QKV, nullptr, 4096, 3072, 1024, 1024, 1024, 3072,
                0, 0, 0, 0, 0, 0, 1, 1, 0, 0, 0, 0, 0, 1.f, 0.f);
    k_scale_q<<<16384, 256, 0, stream>>>(QKV);
    k_landmark<<<4096, 256, 0, stream>>>(QKV, QL, 0);
    k_landmark<<<4096, 256, 0, stream>>>(QKV, KL, 1024);
    // a1 = q @ kl^T
    gemm_launch(stream, QKV, KL, A1, nullptr, 2048, 512, 128, 3072, 128, 512,
                SQB, 128, SLMB, SLM, SA1B, SA1, 8, 16, 0, 0, 0, 0, 0, 1.f, 0.f);
    k_softmax<<<32768, 256, 0, stream>>>(A1, 512);
    // a2 = ql @ kl^T
    gemm_launch(stream, QL, KL, A2, nullptr, 512, 512, 128, 128, 128, 512,
                SLMB, SLM, SLMB, SLM, SA2B, SA2, 8, 16, 0, 0, 0, 0, 0, 1.f, 0.f);
    k_softmax<<<8192, 256, 0, stream>>>(A2, 512);
    // a3 = ql @ k^T
    gemm_launch(stream, QL, QKV + 1024, A3, nullptr, 512, 2048, 128, 128, 3072, 2048,
                SLMB, SLM, SQB, 128, SA3B, SA3, 8, 16, 0, 0, 0, 0, 0, 1.f, 0.f);
    k_softmax<<<8192, 256, 0, stream>>>(A3, 2048);
    // pinv(a2)
    k_pinv_rowcol<<<16, 256, 0, stream>>>(A2, RED);
    { dim3 gz(16, 64); k_pinv_make_z<<<gz, 256, 0, stream>>>(A2, RED, ZB); }
    float* zc = ZB; float* zo = Z2;
    for (int it = 0; it < 6; ++it) {
      gemm_launch(stream, A2, zc, PB, nullptr, 512, 512, 512, 512, 512, 512,
                  SA2B, SA2, SA2B, SA2, SA2B, SA2, 8, 16, 1, 0, 0, 0, 0, 1.f, 0.f);   // P = a2@z
      k_diag_lin<<<16384, 256, 0, stream>>>(PB, W1, -1.f, 7.f);                        // Q = 7I-P
      gemm_launch(stream, PB, W1, W2, nullptr, 512, 512, 512, 512, 512, 512,
                  SA2B, SA2, SA2B, SA2, SA2B, SA2, 8, 16, 1, 0, 0, 0, 0, -1.f, 15.f);  // S = 15I-P@Q
      gemm_launch(stream, PB, W2, W1, nullptr, 512, 512, 512, 512, 512, 512,
                  SA2B, SA2, SA2B, SA2, SA2B, SA2, 8, 16, 1, 0, 0, 0, 0, -1.f, 13.f);  // U = 13I-P@S
      gemm_launch(stream, zc, W1, zo, nullptr, 512, 512, 512, 512, 512, 512,
                  SA2B, SA2, SA2B, SA2, SA2B, SA2, 8, 16, 1, 0, 0, 0, 0, 0.25f, 0.f);  // z' = .25 z@U
      float* t = zc; zc = zo; zo = t;
    }
    // a3v = a3 @ v
    gemm_launch(stream, A3, QKV + 2048, A3V, nullptr, 512, 128, 2048, 2048, 3072, 128,
                SA3B, SA3, SQB, 128, 8 * SLM, SLM, 8, 16, 1, 0, 0, 0, 0, 1.f, 0.f);
    // t2 = pinv @ a3v
    gemm_launch(stream, zc, A3V, T2, nullptr, 512, 128, 512, 512, 128, 128,
                SA2B, SA2, 8 * SLM, SLM, 8 * SLM, SLM, 8, 16, 1, 0, 0, 0, 0, 1.f, 0.f);
    // att = a1 @ t2   (written straight into [b][t][h*128+d])
    gemm_launch(stream, A1, T2, ATT, nullptr, 2048, 128, 512, 512, 128, 1024,
                SA1B, SA1, 8 * SLM, SLM, 2048LL * 1024, 128, 8, 16, 1, 0, 0, 0, 0, 1.f, 0.f);
    // depthwise 33-tap residual conv on v, added into att
    k_resconv<<<16384, 256, 0, stream>>>(QKV, resw, ATT);
    // output projection, accumulated into H (residual)
    gemm_launch(stream, ATT, outw, H, outb, 4096, 1024, 1024, 1024, 1024, 1024,
                0, 0, 0, 0, 0, 0, 1, 1, 0, 0, 0, /*accum*/1, 0, 1.f, 0.f);
  };

  attn_layer(ln1_g, ln1_b, qkv1_w, out1_w, out1_b, res1_w);

  // ---- PPEG (three sequential residual depthwise convs) ----
  k_ppeg<<<16384, 256, 0, stream>>>(H, A1, ppeg_w7, ppeg_b7, 7);
  k_ppeg<<<16384, 256, 0, stream>>>(A1, A3, ppeg_w5, ppeg_b5, 5);
  k_ppeg<<<16384, 256, 0, stream>>>(A3, H, ppeg_w3, ppeg_b3, 3);

  attn_layer(ln2_g, ln2_b, qkv2_w, out2_w, out2_b, res2_w);

  // ---- final norm + head ----
  k_layernorm<<<4096, 256, 0, stream>>>(H, XL, normf_g, normf_b, 1024);
  k_head<<<4, 256, 0, stream>>>(XL, fc2_w, fc2_b, (float*)d_out);
}
